// Attention_83537113907346
// MI455X (gfx1250) — matrix operand order repaired and checked by reading, not compiled, not run on hardware
//
#include <hip/hip_runtime.h>
#include <hip/hip_bf16.h>

typedef __attribute__((ext_vector_type(16))) _Float16 v16h;
typedef __attribute__((ext_vector_type(8)))  _Float16 v8h;
typedef __attribute__((ext_vector_type(8)))  float    v8f;

#define NHEADS 8
#define HD     64
#define SEQ    4096
#define DIM    512

static __device__ __forceinline__ int uni(int v) {
    return __builtin_amdgcn_readfirstlane(v);   // force wave-uniform (scalar) value
}

// ---------------------------------------------------------------------------
// Kernel 1: fused QKV projection.  out = x @ [Wq;Wkv]^T  (N columns 0..1535)
// cols    0..511  -> Q stored [h][n][64] f16
// cols  512..1023 -> K stored [h][n][64] f16
// cols 1024..1535 -> V stored TRANSPOSED [h][64][n] f16
// One wave computes one 16x16 tile via v_wmma_f32_16x16x32_f16 (K=512 loop).
// ---------------------------------------------------------------------------
__global__ void qkv_proj(const float* __restrict__ x,
                         const float* __restrict__ Wq,
                         const float* __restrict__ Wkv,
                         _Float16* __restrict__ Qh,
                         _Float16* __restrict__ Kh,
                         _Float16* __restrict__ Vt) {
    const int lane = threadIdx.x & 31;
    const int tile = uni(blockIdx.x * 8 + (threadIdx.x >> 5));   // 24576 tiles
    const int m0 = uni((tile / 96) << 4);                        // 256 row tiles
    const int n0 = uni((tile % 96) << 4);                        // 96 col tiles
    const int lh = lane >> 4;
    const int ll = lane & 15;

    const float* W;
    int nloc;
    if (n0 < DIM) { W = Wq;  nloc = n0; }            // scalar branch
    else          { W = Wkv; nloc = n0 - DIM; }

    const int m  = m0 + ll;
    const int nl = nloc + ll;

    v8f zero = {};
    v8f c = zero;

    for (int kc = 0; kc < DIM; kc += 32) {
        v16h a, b;
        {   // A tile: row = ll ; K = kb..kb+7 and kb+16..kb+23, kb = lh*8
            const int kb = kc + lh * 8;
            v8f x0 = *(const v8f*)(x + (size_t)m * DIM + kb);
            v8f x1 = *(const v8f*)(x + (size_t)m * DIM + kb + 16);
#pragma unroll
            for (int i = 0; i < 8; ++i) { a[i] = (_Float16)x0[i]; a[8 + i] = (_Float16)x1[i]; }
        }
        {   // B tile: col = ll ; K = wk..wk+7 and wk+16..wk+23, wk = lh*8  (B[k][n] = W[n][k])
            const int wk = kc + lh * 8;
            v8f w0 = *(const v8f*)(W + (size_t)nl * DIM + wk);
            v8f w1 = *(const v8f*)(W + (size_t)nl * DIM + wk + 16);
#pragma unroll
            for (int i = 0; i < 8; ++i) { b[i] = (_Float16)w0[i]; b[8 + i] = (_Float16)w1[i]; }
        }
        c = __builtin_amdgcn_wmma_f32_16x16x32_f16(false, a, false, b, (short)0, c, false, false);
    }

    // scatter D (C layout: row = e + lh*8, col = ll); region branch is scalar
    if (n0 < DIM) {
        const int hh = (n0 + ll) >> 6, dd = (n0 + ll) & 63;
#pragma unroll
        for (int e = 0; e < 8; ++e)
            Qh[((size_t)hh * SEQ + (m0 + e + lh * 8)) * HD + dd] = (_Float16)c[e];
    } else if (n0 < 2 * DIM) {
        const int c2 = n0 - DIM + ll, hh = c2 >> 6, dd = c2 & 63;
#pragma unroll
        for (int e = 0; e < 8; ++e)
            Kh[((size_t)hh * SEQ + (m0 + e + lh * 8)) * HD + dd] = (_Float16)c[e];
    } else {
        const int c2 = n0 - 2 * DIM + ll, hh = c2 >> 6, dd = c2 & 63;
#pragma unroll
        for (int e = 0; e < 8; ++e)
            Vt[((size_t)hh * HD + dd) * SEQ + (m0 + e + lh * 8)] = (_Float16)c[e]; // transposed
    }
}

// ---------------------------------------------------------------------------
// Kernel 2: flash attention.  One wave = one (head, 16-query) tile.
// Streams keys in blocks of 64 (4 S-tiles) with online masked softmax;
// the cross-lane row-max/row-sum reductions are amortized over 64 keys.
// mask = 1 - eye: diagonal contributes 0 to the running max and 0 to the sum.
// ---------------------------------------------------------------------------
__global__ void attn_flash(const _Float16* __restrict__ Qh,
                           const _Float16* __restrict__ Kh,
                           const _Float16* __restrict__ Vt,
                           const float* __restrict__ scale_p,
                           _Float16* __restrict__ O) {
    __shared__ _Float16 plds[8][16 * 64];   // per-wave P transpose staging (16 KB)
    const int lane = threadIdx.x & 31;
    const int wave = threadIdx.x >> 5;
    const int tile = uni(blockIdx.x * 8 + wave);     // 2048 tiles
    const int h  = uni(tile >> 8);                   // 256 q-tiles per head
    const int q0 = uni((tile & 255) << 4);
    const float scale = scale_p[0];
    const int lh = lane >> 4;
    const int ll = lane & 15;

    // ---- load Q A-tiles (16 rows x 64) once ----
    const _Float16* qb = Qh + ((size_t)h * SEQ + q0) * HD;
    v16h aq0, aq1;
    {
        const int kb = lh * 16;
        v8h q_0 = *(const v8h*)(qb + ll * HD + kb);
        v8h q_1 = *(const v8h*)(qb + ll * HD + kb + 8);
        v8h q_2 = *(const v8h*)(qb + ll * HD + kb + 32);
        v8h q_3 = *(const v8h*)(qb + ll * HD + kb + 40);
#pragma unroll
        for (int i = 0; i < 8; ++i) {
            aq0[i] = q_0[i]; aq0[8 + i] = q_1[i];
            aq1[i] = q_2[i]; aq1[8 + i] = q_3[i];
        }
    }

    float m_run[8], l_run[8];
#pragma unroll
    for (int e = 0; e < 8; ++e) { m_run[e] = 0.0f; l_run[e] = 0.0f; }
    v8f zero = {};
    v8f acc[4] = {zero, zero, zero, zero};

    const _Float16* kb_ = Kh + (size_t)h * SEQ * HD;
    const _Float16* vb_ = Vt + (size_t)h * HD * SEQ;
    _Float16* pt = plds[wave];

    for (int j = 0; j < SEQ; j += 64) {
        if (j + 64 < SEQ) {
            __builtin_prefetch(kb_ + (size_t)(j + 64 + ll + lh * 16) * HD, 0, 1);
            __builtin_prefetch(vb_ + (size_t)(ll + lh * 16) * SEQ + j + 64, 0, 1);
        }

        // ---- S = Q K^T for 4 key sub-blocks of 16 ----
        v8f s[4] = {zero, zero, zero, zero};
#pragma unroll
        for (int t = 0; t < 4; ++t) {
            const _Float16* kr = kb_ + (size_t)(j + t * 16 + ll) * HD + lh * 16;
            v16h bk0 = *(const v16h*)(kr);
            v16h bk1 = *(const v16h*)(kr + 32);
            s[t] = __builtin_amdgcn_wmma_f32_16x16x32_f16(false, aq0, false, bk0, (short)0, s[t], false, false);
            s[t] = __builtin_amdgcn_wmma_f32_16x16x32_f16(false, aq1, false, bk1, (short)0, s[t], false, false);
        }

        // ---- online masked softmax update (per row = e + lh*8) ----
        float fac[8];
#pragma unroll
        for (int e = 0; e < 8; ++e) {
            const int grow = q0 + e + lh * 8;
            const int row  = e + lh * 8;
            float av[4];
#pragma unroll
            for (int t = 0; t < 4; ++t) {
                float a = s[t][e] * scale;
                if (grow == j + t * 16 + ll) a = 0.0f;      // masked_vec = s * mask
                av[t] = a;
            }
            float bm = fmaxf(fmaxf(av[0], av[1]), fmaxf(av[2], av[3]));
#pragma unroll
            for (int d = 1; d < 16; d <<= 1) bm = fmaxf(bm, __shfl_xor(bm, d, 32));
            const float mn = fmaxf(m_run[e], bm);
            fac[e] = __expf(m_run[e] - mn);
            float rs = 0.0f;
#pragma unroll
            for (int t = 0; t < 4; ++t) {
                float ev = __expf(av[t] - mn);
                if (grow == j + t * 16 + ll) ev = 0.0f;     // exps * mask
                rs += ev;
                pt[row * 64 + t * 16 + ll] = (_Float16)ev;  // pack P straight into LDS
            }
#pragma unroll
            for (int d = 1; d < 16; d <<= 1) rs += __shfl_xor(rs, d, 32);
            l_run[e] = l_run[e] * fac[e] + rs;
            m_run[e] = mn;
        }

        // ---- rescale accumulators ----
#pragma unroll
        for (int t = 0; t < 4; ++t)
#pragma unroll
            for (int e = 0; e < 8; ++e) acc[t][e] *= fac[e];

        // ---- reload P in A layout (16x64 -> two 16x32 A tiles) ----
        v16h ap0, ap1;
        {
            const int kb = lh * 16;
            const _Float16* base = pt + ll * 64;
            v8h x0 = *(const v8h*)(base + kb);
            v8h x1 = *(const v8h*)(base + 8 + kb);
            v8h x2 = *(const v8h*)(base + 32 + kb);
            v8h x3 = *(const v8h*)(base + 40 + kb);
#pragma unroll
            for (int i = 0; i < 8; ++i) {
                ap0[i] = x0[i]; ap0[8 + i] = x1[i];
                ap1[i] = x2[i]; ap1[8 + i] = x3[i];
            }
        }

        // ---- O += P V  (V transposed: B[k][n] contiguous along keys) ----
#pragma unroll
        for (int t = 0; t < 4; ++t) {
            const _Float16* vr = vb_ + (size_t)(t * 16 + ll) * SEQ + j + lh * 16;
            v16h bv0 = *(const v16h*)(vr);
            v16h bv1 = *(const v16h*)(vr + 32);
            acc[t] = __builtin_amdgcn_wmma_f32_16x16x32_f16(false, ap0, false, bv0, (short)0, acc[t], false, false);
            acc[t] = __builtin_amdgcn_wmma_f32_16x16x32_f16(false, ap1, false, bv1, (short)0, acc[t], false, false);
        }
    }

    // ---- normalize and store O[n][c] f16, c = h*64 + d ----
#pragma unroll
    for (int e = 0; e < 8; ++e) {
        float l = l_run[e]; if (l == 0.0f) l = 1.0f;   // masked_sums + (==0)
        const float inv = 1.0f / l;
        const int row = q0 + e + lh * 8;
#pragma unroll
        for (int t = 0; t < 4; ++t)
            O[(size_t)row * DIM + h * HD + t * 16 + ll] = (_Float16)(acc[t][e] * inv);
    }
}

// ---------------------------------------------------------------------------
// Kernel 3: output projection  out = O @ Wproj^T + bproj   (fp32 out)
// ---------------------------------------------------------------------------
__global__ void out_proj(const _Float16* __restrict__ O,
                         const float* __restrict__ Wp,
                         const float* __restrict__ bp,
                         float* __restrict__ out) {
    const int lane = threadIdx.x & 31;
    const int tile = uni(blockIdx.x * 8 + (threadIdx.x >> 5));   // 8192 tiles
    const int m0 = uni((tile / 32) << 4);
    const int n0 = uni((tile % 32) << 4);
    const int lh = lane >> 4;
    const int ll = lane & 15;

    const int m  = m0 + ll;
    const int nl = n0 + ll;
    v8f zero = {};
    v8f c = zero;

    for (int kc = 0; kc < DIM; kc += 32) {
        v16h a, b;
        {
            const int kb = kc + lh * 8;
            v8h a0 = *(const v8h*)(O + (size_t)m * DIM + kb);
            v8h a1 = *(const v8h*)(O + (size_t)m * DIM + kb + 16);
#pragma unroll
            for (int i = 0; i < 8; ++i) { a[i] = a0[i]; a[8 + i] = a1[i]; }
        }
        {
            const int wk = kc + lh * 8;
            v8f w0 = *(const v8f*)(Wp + (size_t)nl * DIM + wk);
            v8f w1 = *(const v8f*)(Wp + (size_t)nl * DIM + wk + 16);
#pragma unroll
            for (int i = 0; i < 8; ++i) { b[i] = (_Float16)w0[i]; b[8 + i] = (_Float16)w1[i]; }
        }
        c = __builtin_amdgcn_wmma_f32_16x16x32_f16(false, a, false, b, (short)0, c, false, false);
    }

    const float bias = bp[n0 + ll];
#pragma unroll
    for (int e = 0; e < 8; ++e) {
        const int row = m0 + e + lh * 8;
        out[(size_t)row * DIM + n0 + ll] = c[e] + bias;
    }
}

// ---------------------------------------------------------------------------
extern "C" void kernel_launch(void* const* d_in, const int* in_sizes, int n_in,
                              void* d_out, int out_size, void* d_ws, size_t ws_size,
                              hipStream_t stream) {
    // inputs: x, H, W, scale, Wq, Wkv, Wproj, bproj
    const float* x     = (const float*)d_in[0];
    const float* scale = (const float*)d_in[3];
    const float* Wq    = (const float*)d_in[4];
    const float* Wkv   = (const float*)d_in[5];
    const float* Wp    = (const float*)d_in[6];
    const float* bpj   = (const float*)d_in[7];
    float* out = (float*)d_out;

    char* ws = (char*)d_ws;
    const size_t seg = (size_t)NHEADS * SEQ * HD * sizeof(_Float16); // 4 MB
    _Float16* Qh = (_Float16*)(ws);
    _Float16* Kh = (_Float16*)(ws + seg);
    _Float16* Vt = (_Float16*)(ws + 2 * seg);
    _Float16* O  = (_Float16*)(ws + 3 * seg);

    qkv_proj  <<<3072, 256, 0, stream>>>(x, Wq, Wkv, Qh, Kh, Vt);
    attn_flash<<< 256, 256, 0, stream>>>(Qh, Kh, Vt, scale, O);
    out_proj  <<<1024, 256, 0, stream>>>(O, Wp, bpj, out);
    (void)in_sizes; (void)n_in; (void)out_size; (void)ws_size;
}
